// MultiHeadAttention_17738214933247
// MI455X (gfx1250) — matrix operand order repaired and checked by reading, not compiled, not run on hardware
//
#include <hip/hip_runtime.h>

typedef _Float16 half_t;
typedef __attribute__((ext_vector_type(16))) _Float16 v16h;
typedef __attribute__((ext_vector_type(8)))  _Float16 v8h;
typedef __attribute__((ext_vector_type(8)))  float    v8f;

union V16 { v16h v; v8h h[2]; };

#define SEQ    2048
#define BATCH  2
#define NROWS  4096          // BATCH*SEQ
#define HID    1024
#define NHEAD  16
#define DK     64

static __device__ __forceinline__ v8f wmma16(v16h a, v16h b, v8f c) {
  // D = A(16x32 f16) x B(32x16 f16) + C(16x16 f32)
  return __builtin_amdgcn_wmma_f32_16x16x32_f16(false, a, false, b, (short)0, c,
                                                false, false);
}

// ---------------------------------------------------------------------------
// Tiled GEMM: Y[M,1024] = X[M,1024] @ W[1024,1024]^T  (torch Linear convention)
// Block tile 128x128, k-step 32, 8 waves (2M x 4N), each wave 64x32 = 8 frags.
// MODE 0: f32 input, epilogue writes f16 into WMMA-fragment-swizzled Q/K/V
//         layouts (z = 0/1/2 selects mapping; Q pre-scaled by 1/sqrt(64)).
// MODE 1: f16 input, epilogue writes f32 row-major (final projection).
// ---------------------------------------------------------------------------
template <int MODE>
__global__ __launch_bounds__(256) void gemm_xwt(
    const void* x0, const void* x1, const void* x2,
    const float* w0, const float* w1, const float* w2,
    void* y0, void* y1, void* y2)
{
  __shared__ half_t As[128][40];   // +8 halfs pad -> 20-bank row stride
  __shared__ half_t Bs[128][40];

  const void*  xs[3]   = {x0, x1, x2};
  const float* wsel[3] = {w0, w1, w2};
  void*        ys[3]   = {y0, y1, y2};
  const int z = blockIdx.z;
  const float* Wp = wsel[z];

  const int tid   = threadIdx.x;
  const int lane  = tid & 31;
  const int wv    = tid >> 5;
  const int m15   = lane & 15;
  const int hi    = lane >> 4;
  const int waveM = wv >> 2;          // 0..1
  const int waveN = wv & 3;           // 0..3
  const int ldRow = tid >> 1;         // 0..127
  const int ldCol = (tid & 1) * 16;   // 0 or 16

  const size_t rowA0 = (size_t)blockIdx.y * 128;
  const size_t rowB0 = (size_t)blockIdx.x * 128;

  v8f zf = {};
  v8f acc[4][2];
#pragma unroll
  for (int i = 0; i < 4; ++i)
#pragma unroll
    for (int j = 0; j < 2; ++j) acc[i][j] = zf;

  for (int kk = 0; kk < HID / 32; ++kk) {
    // ---- stage A tile (convert f32->f16 if needed) ----
    if (MODE == 1) {
      const half_t* X = (const half_t*)xs[z];
      const half_t* src = X + (rowA0 + ldRow) * HID + kk * 32 + ldCol;
      *(v8h*)&As[ldRow][ldCol]     = *(const v8h*)(src);
      *(v8h*)&As[ldRow][ldCol + 8] = *(const v8h*)(src + 8);
      if (kk + 1 < HID / 32) __builtin_prefetch(src + 32, 0, 3);
    } else {
      const float* X = (const float*)xs[z];
      const float* src = X + (rowA0 + ldRow) * HID + kk * 32 + ldCol;
      v8h h0, h1;
#pragma unroll
      for (int i = 0; i < 8; ++i) { h0[i] = (half_t)src[i]; h1[i] = (half_t)src[8 + i]; }
      *(v8h*)&As[ldRow][ldCol]     = h0;
      *(v8h*)&As[ldRow][ldCol + 8] = h1;
      if (kk + 1 < HID / 32) __builtin_prefetch(src + 32, 0, 3);
    }
    // ---- stage B tile (weights, always f32 source) ----
    {
      const float* src = Wp + (rowB0 + ldRow) * HID + kk * 32 + ldCol;
      v8h h0, h1;
#pragma unroll
      for (int i = 0; i < 8; ++i) { h0[i] = (half_t)src[i]; h1[i] = (half_t)src[8 + i]; }
      *(v8h*)&Bs[ldRow][ldCol]     = h0;
      *(v8h*)&Bs[ldRow][ldCol + 8] = h1;
      if (kk + 1 < HID / 32) __builtin_prefetch(src + 32, 0, 3);
    }
    __syncthreads();

    // ---- fragments per documented 16-bit WMMA lane layouts ----
    V16 aF[4], bF[2];
#pragma unroll
    for (int fm = 0; fm < 4; ++fm) {
      const int m = waveM * 64 + fm * 16 + m15;
      aF[fm].h[0] = *(const v8h*)&As[m][hi * 8];        // K = hi*8 .. +7
      aF[fm].h[1] = *(const v8h*)&As[m][16 + hi * 8];   // K = 16+hi*8 .. +7
    }
#pragma unroll
    for (int fn = 0; fn < 2; ++fn) {
      const int n = waveN * 32 + fn * 16 + m15;
      bF[fn].h[0] = *(const v8h*)&Bs[n][hi * 8];        // K = hi*8 .. +7
      bF[fn].h[1] = *(const v8h*)&Bs[n][16 + hi * 8];   // K = 16+hi*8 .. +7
    }
#pragma unroll
    for (int fm = 0; fm < 4; ++fm)
#pragma unroll
      for (int fn = 0; fn < 2; ++fn)
        acc[fm][fn] = wmma16(aF[fm].v, bF[fn].v, acc[fm][fn]);
    __syncthreads();
  }

  // ---- epilogue: C layout lane(r): M = r + hi*8, N = m15 ----
#pragma unroll
  for (int fm = 0; fm < 4; ++fm) {
    const int row0 = (int)rowA0 + waveM * 64 + fm * 16 + hi * 8;
#pragma unroll
    for (int fn = 0; fn < 2; ++fn) {
      const int col = (int)rowB0 + waveN * 32 + fn * 16 + m15;
#pragma unroll
      for (int r = 0; r < 8; ++r) {
        const float vsc = acc[fm][fn][r];
        if (MODE == 1) {
          ((float*)ys[z])[(size_t)(row0 + r) * HID + col] = vsc;
        } else {
          // fused fragment swizzle, replaces the reshape pass
          const int rowg = row0 + r;
          const int b  = rowg >> 11;          // /SEQ
          const int s  = rowg & (SEQ - 1);
          const int h  = col >> 6;            // /DK
          const int dd = col & (DK - 1);
          const int bh = b * NHEAD + h;
          half_t* Y = (half_t*)ys[z];
          if (z == 0) {          // Q: A-frag tiles 16(s) x 32(d), scaled
            const int td = dd >> 5, k = dd & 31;
            const int ts = s >> 4, m = s & 15;
            const int laneA = m | ((k & 8) << 1);
            const int ii = (k & 7) + ((k & 16) >> 1);
            Y[((size_t)((bh * 128 + ts) * 2 + td)) * 512 + laneA * 16 + ii] =
                (half_t)(vsc * 0.125f);
          } else if (z == 1) {   // K^T: B-frag tiles 32(d) x 16(s)
            const int td = dd >> 5, K = dd & 31;
            const int tk = s >> 4, N = s & 15;
            const int laneB = N | ((K & 8) << 1);
            const int ii = (K & 7) + ((K & 16) >> 1);
            Y[((size_t)((bh * 128 + tk) * 2 + td)) * 512 + laneB * 16 + ii] =
                (half_t)vsc;
          } else {               // V: B-frag tiles 32(s) x 16(d)
            const int tk = s >> 5, K = s & 31;
            const int td = dd >> 4, N = dd & 15;
            const int laneB = N | ((K & 8) << 1);
            const int ii = (K & 7) + ((K & 16) >> 1);
            Y[((size_t)((bh * 64 + tk) * 4 + td)) * 512 + laneB * 16 + ii] =
                (half_t)vsc;
          }
        }
      }
    }
  }
}

// ---------------------------------------------------------------------------
// Flash attention: block = 8 waves; each wave owns 16 q-rows, streams kv in
// steps of 32. QK^T (4 wmma) -> mask -> online softmax (shfl_xor half-wave
// reductions) -> P to LDS in A-fragment order -> PV (4 wmma).
// ---------------------------------------------------------------------------
__global__ __launch_bounds__(256) void attn_fwd(
    const half_t* __restrict__ Qs, const half_t* __restrict__ Ks,
    const half_t* __restrict__ Vs, const unsigned char* __restrict__ mask,
    half_t* __restrict__ Oh)
{
  __shared__ half_t Pbuf[8][512];     // per-wave 16x32 P tile, A-frag order

  const int tid  = threadIdx.x;
  const int lane = tid & 31;
  const int wv   = tid >> 5;
  const int m15  = lane & 15;
  const int hi   = lane >> 4;
  const int bh   = blockIdx.y;
  const int b    = bh >> 4;
  const int h    = bh & 15;
  const int qb   = blockIdx.x * 128 + wv * 16;

  // Q fragments: 16 rows x 64 d = two A-frags, kept in registers all loop
  V16 aQ[2];
#pragma unroll
  for (int td = 0; td < 2; ++td) {
    const half_t* p = Qs + ((size_t)((bh * 128 + (qb >> 4)) * 2 + td)) * 512 + lane * 16;
    aQ[td].h[0] = *(const v8h*)p;
    aQ[td].h[1] = *(const v8h*)(p + 8);
  }

  v8f zf = {};
  v8f acc[4];                          // O accumulator: 16 rows x 64 d
#pragma unroll
  for (int i = 0; i < 4; ++i) acc[i] = zf;
  float mrow[8], lrow[8];
#pragma unroll
  for (int r = 0; r < 8; ++r) { mrow[r] = -__builtin_inff(); lrow[r] = 0.f; }

  const unsigned char* mbase =
      mask + ((size_t)b * SEQ + qb + hi * 8) * SEQ + m15;

  for (int t = 0; t < SEQ / 32; ++t) {
    // ---- scores: two 16x16 C tiles over kv cols [t*32, t*32+32) ----
    v8f sc[2];
#pragma unroll
    for (int j = 0; j < 2; ++j) {
      const int tk16 = 2 * t + j;
      V16 bK0, bK1;
      const half_t* p0 = Ks + ((size_t)((bh * 128 + tk16) * 2 + 0)) * 512 + lane * 16;
      const half_t* p1 = Ks + ((size_t)((bh * 128 + tk16) * 2 + 1)) * 512 + lane * 16;
      bK0.h[0] = *(const v8h*)p0; bK0.h[1] = *(const v8h*)(p0 + 8);
      bK1.h[0] = *(const v8h*)p1; bK1.h[1] = *(const v8h*)(p1 + 8);
      v8f s = wmma16(aQ[0].v, bK0.v, zf);
      sc[j]  = wmma16(aQ[1].v, bK1.v, s);
    }
    // ---- mask (bool, row = qb+r+hi*8, col = t*32 + j*16 + m15) ----
    const unsigned char* mp = mbase + (size_t)t * 32;
#pragma unroll
    for (int r = 0; r < 8; ++r) {
      if (mp[(size_t)r * SEQ])      sc[0][r] = -1e9f;
      if (mp[(size_t)r * SEQ + 16]) sc[1][r] = -1e9f;
    }
    // ---- online softmax; row values live in 16 lanes of one half-wave ----
#pragma unroll
    for (int r = 0; r < 8; ++r) {
      float mx = fmaxf(sc[0][r], sc[1][r]);
#pragma unroll
      for (int off = 8; off >= 1; off >>= 1)
        mx = fmaxf(mx, __shfl_xor(mx, off, 32));
      const float mnew = fmaxf(mrow[r], mx);
      const float p0 = __expf(sc[0][r] - mnew);
      const float p1 = __expf(sc[1][r] - mnew);
      float sum = p0 + p1;
#pragma unroll
      for (int off = 8; off >= 1; off >>= 1)
        sum += __shfl_xor(sum, off, 32);
      const float alpha = __expf(mrow[r] - mnew);
      lrow[r] = lrow[r] * alpha + sum;
      mrow[r] = mnew;
#pragma unroll
      for (int i = 0; i < 4; ++i) acc[i][r] *= alpha;
      // scatter P element (M=r+hi*8, k=j*16+m15) into A-fragment order
      const int laneA = (r + hi * 8) + ((m15 & 8) << 1);
      const int ib = m15 & 7;
      Pbuf[wv][laneA * 16 + ib]     = (half_t)p0;
      Pbuf[wv][laneA * 16 + ib + 8] = (half_t)p1;
    }
    // same-wave DS ops are in-order: safe to read back immediately
    V16 aP;
    aP.h[0] = *(const v8h*)&Pbuf[wv][lane * 16];
    aP.h[1] = *(const v8h*)&Pbuf[wv][lane * 16 + 8];
    // ---- PV: P(16x32) @ V(32x64) ----
#pragma unroll
    for (int td2 = 0; td2 < 4; ++td2) {
      V16 bV;
      const half_t* p = Vs + ((size_t)((bh * 64 + t) * 4 + td2)) * 512 + lane * 16;
      bV.h[0] = *(const v8h*)p; bV.h[1] = *(const v8h*)(p + 8);
      acc[td2] = wmma16(aP.v, bV.v, acc[td2]);
    }
  }

  // ---- normalize and store O in [b, s, h*64+d] f16 for final projection ----
  float inv[8];
#pragma unroll
  for (int r = 0; r < 8; ++r) inv[r] = lrow[r] > 0.f ? 1.f / lrow[r] : 0.f;
#pragma unroll
  for (int td2 = 0; td2 < 4; ++td2)
#pragma unroll
    for (int r = 0; r < 8; ++r) {
      const int row = qb + r + hi * 8;
      const int col = h * DK + td2 * 16 + m15;
      Oh[((size_t)b * SEQ + row) * HID + col] = (half_t)(acc[td2][r] * inv[r]);
    }
}

// ---------------------------------------------------------------------------
extern "C" void kernel_launch(void* const* d_in, const int* in_sizes, int n_in,
                              void* d_out, int out_size, void* d_ws, size_t ws_size,
                              hipStream_t stream) {
  (void)in_sizes; (void)n_in; (void)out_size; (void)ws_size;
  const float* q  = (const float*)d_in[0];
  const float* k  = (const float*)d_in[1];
  const float* v  = (const float*)d_in[2];
  const unsigned char* mask = (const unsigned char*)d_in[3];
  const float* Wq = (const float*)d_in[4];
  const float* Wk = (const float*)d_in[5];
  const float* Wv = (const float*)d_in[6];
  const float* Wo = (const float*)d_in[7];
  float* out = (float*)d_out;

  half_t* ws = (half_t*)d_ws;
  const size_t MAT = (size_t)NROWS * HID;   // 4M halfs = 8MB
  half_t* Qs = ws;
  half_t* Ks = ws + MAT;
  half_t* Vs = ws + 2 * MAT;
  half_t* Oh = ws + 3 * MAT;

  dim3 blk(256);
  // Q/K/V projections (f32 in -> swizzled f16 out), one z-slice per matrix
  gemm_xwt<0><<<dim3(8, 32, 3), blk, 0, stream>>>(
      q, k, v, Wq, Wk, Wv, Qs, Ks, Vs);
  // flash attention
  attn_fwd<<<dim3(16, 32), blk, 0, stream>>>(Qs, Ks, Vs, mask, Oh);
  // output projection (f16 in -> f32 out)
  gemm_xwt<1><<<dim3(8, 32, 1), blk, 0, stream>>>(
      Oh, Oh, Oh, Wo, Wo, Wo, out, out, out);
}
